// EGCL_Multi_49641232007271
// MI455X (gfx1250) — matrix operand order repaired and checked by reading, not compiled, not run on hardware
//
#include <hip/hip_runtime.h>
#include <hip/hip_bf16.h>

typedef _Float16 h16;
typedef __attribute__((ext_vector_type(16))) _Float16 v16h;
typedef __attribute__((ext_vector_type(8)))  _Float16 v8h;
typedef __attribute__((ext_vector_type(4)))  _Float16 v4h;
typedef __attribute__((ext_vector_type(8)))  float    v8f;

#define NNODE 768
#define NHEAD 4
#define DHDIM 64
#define MDIM  128
#define FEP   192   // padded phi_e input: [sq 0..3 | pad 4..7 | feat_j 8..87 | feat_i 88..167 | pad 168..191]
#define PHK   192   // phi_h input 128+64 = 192

__device__ __forceinline__ float fastrcp(float v){ return __builtin_amdgcn_rcpf(v); }
__device__ __forceinline__ float siluf(float v){ return v * fastrcp(1.f + __expf(-v)); }
__device__ __forceinline__ float sigm (float v){ return fastrcp(1.f + __expf(-v)); }

__device__ __forceinline__ v8f wmma16(v16h a, v16h b, v8f c){
  // D = A(16x32 f16) * B(32x16 f16) + C(f32)
  return __builtin_amdgcn_wmma_f32_16x16x32_f16(false, a, false, b, (short)0, c, false, false);
}

// A fragment from row-major 16xK LDS tile (16x32 A operand, contiguous k order matching ldB):
// lane l: M=l%16; 16 contiguous K halves starting at k0 (+16 if l>=16).
__device__ __forceinline__ v16h ldA(const h16* t, int lda, int k0, int lane){
  int m   = lane & 15;
  int off = (lane & 16) ? 16 : 0;
  const h16* p = t + m*lda + k0 + off;
  return *(const v16h*)(p);
}

// B fragment (32x16) from K-major (transposed) weight Wt[N][ldk]:
// lane l: N=l%16; 16 contiguous K halves starting at k0 (+16 if l>=16).
__device__ __forceinline__ v16h ldB(const h16* Wt, int ldk, int n0, int k0, int lane){
  int nn = lane & 15;
  int kk = k0 + ((lane & 16) ? 16 : 0);
  return *(const v16h*)(Wt + (size_t)(n0 + nn)*ldk + kk);
}

// One MLP layer: O[16][128] = (act?silu:id)(A[16][K] @ W + bias), W given K-major.
__device__ __forceinline__ void mlp_layer(const h16* A, int lda, int K,
                                          const h16* Wt, const float* __restrict__ bias,
                                          h16* O, int lane, bool act){
  const int nkt = K >> 5;
  const int nn  = lane & 15;
  const int mb  = (lane & 16) ? 8 : 0;
  for (int nt = 0; nt < 8; ++nt){
    v8f c = {};
    for (int kt = 0; kt < nkt; ++kt)
      c = wmma16(ldA(A, lda, kt*32, lane), ldB(Wt, K, nt*16, kt*32, lane), c);
    float bv = bias[nt*16 + nn];
    #pragma unroll
    for (int r = 0; r < 8; ++r){
      float v = c[r] + bv;
      if (act) v = siluf(v);
      O[(mb + r)*MDIM + nt*16 + nn] = (h16)v;
    }
  }
}

// Single 16-col output tile GEMM (for head/gate projections).
__device__ __forceinline__ v8f head_gemm(const h16* A, int lda, int K,
                                         const h16* Wt, int n0, int lane){
  v8f c = {};
  const int nkt = K >> 5;
  for (int kt = 0; kt < nkt; ++kt)
    c = wmma16(ldA(A, lda, kt*32, lane), ldB(Wt, K, n0, kt*32, lane), c);
  return c;
}

// ---------------- prep: f16 feature table + K-major f16 weights ----------------
__device__ void tpad(const float* W, int K, int Nc, h16* Wt, int Kp, int Nr,
                     int tid, int nthr){
  for (int idx = tid; idx < Nr*Kp; idx += nthr){
    int n = idx / Kp, k = idx % Kp;
    float v = (n < Nc && k < K) ? W[(size_t)k*Nc + n] : 0.f;
    Wt[idx] = (h16)v;
  }
}

__global__ void egcl_prep(const float* __restrict__ x, const float* __restrict__ h,
  const float* We1, const float* We2, const float* Winf,
  const float* Wx1, const float* Wx2, const float* Wx3,
  const float* Wc1, const float* Wc2, const float* Wc3,
  const float* Wh1, const float* Wh2, const float* Wh3,
  h16* feat, h16* We1t, h16* We2t, h16* Winft, h16* Wx1t, h16* Wx2t, h16* Wx3t,
  h16* Wc1t, h16* Wc2t, h16* Wc3t, h16* Wh1t, h16* Wh2t, h16* Wh3t)
{
  int tid  = blockIdx.x*blockDim.x + threadIdx.x;
  int nthr = gridDim.x*blockDim.x;
  // feat[i][0..63]=h, feat[i][64..79]=per-node head-pair squared norms (a*4+b)
  for (int idx = tid; idx < NNODE*80; idx += nthr){
    int i = idx / 80, c = idx % 80;
    float v;
    if (c < 64) v = h[i*64 + c];
    else {
      int q = c - 64; int a = q >> 2, b = q & 3;
      float d0 = x[i*12 + b*3 + 0] - x[i*12 + a*3 + 0];
      float d1 = x[i*12 + b*3 + 1] - x[i*12 + a*3 + 1];
      float d2 = x[i*12 + b*3 + 2] - x[i*12 + a*3 + 2];
      v = d0*d0 + d1*d1 + d2*d2;   // zero on diagonal automatically
    }
    feat[idx] = (h16)v;
  }
  // We1t with padded-K remap: dest k 0..3 <- rows 0..3; 8..167 <- rows 4..163; rest zero
  for (int idx = tid; idx < 128*192; idx += nthr){
    int n = idx / 192, k = idx % 192;
    int src = (k < 4) ? k : ((k >= 8 && k < 168) ? (k - 4) : -1);
    We1t[idx] = (h16)((src >= 0) ? We1[(size_t)src*128 + n] : 0.f);
  }
  tpad(We2, 128, 128, We2t, 128, 128, tid, nthr);
  tpad(Winf,128,   1, Winft, 128,  16, tid, nthr);
  tpad(Wx1, 128, 128, Wx1t, 128, 128, tid, nthr);
  tpad(Wx2, 128, 128, Wx2t, 128, 128, tid, nthr);
  tpad(Wx3, 128,   4, Wx3t, 128,  16, tid, nthr);
  tpad(Wc1, 128, 128, Wc1t, 128, 128, tid, nthr);
  tpad(Wc2, 128, 128, Wc2t, 128, 128, tid, nthr);
  tpad(Wc3, 128,  16, Wc3t, 128,  16, tid, nthr);
  tpad(Wh1, 192, 128, Wh1t, 192, 128, tid, nthr);
  tpad(Wh2, 128, 128, Wh2t, 128, 128, tid, nthr);
  tpad(Wh3, 128,  64, Wh3t, 128,  64, tid, nthr);
}

// ---------------- pass 1: fused all-pairs edge pipeline ----------------
#define EWAVES 4
__global__ void __launch_bounds__(EWAVES*32) egcl_edge(
    const float* __restrict__ x, const h16* __restrict__ feat,
    const h16* __restrict__ We1t, const h16* __restrict__ We2t,
    const h16* __restrict__ Winft, const h16* __restrict__ Wx1t,
    const h16* __restrict__ Wx2t, const h16* __restrict__ Wx3t,
    const float* __restrict__ be1, const float* __restrict__ be2,
    const float* __restrict__ binf, const float* __restrict__ bx1,
    const float* __restrict__ bx2, const float* __restrict__ bx3,
    float* __restrict__ mi, float* __restrict__ Sx)
{
  __shared__ __align__(32) h16 sEin[EWAVES][16][FEP];
  __shared__ __align__(32) h16 sA  [EWAVES][16][MDIM];
  __shared__ __align__(32) h16 sB  [EWAVES][16][MDIM];
  __shared__ float sSq[EWAVES][16][NHEAD];
  __shared__ float sE [EWAVES][16];
  __shared__ float sPx[EWAVES][16][NHEAD];
  __shared__ float sXi[EWAVES][12];

  const int wave = threadIdx.x >> 5;
  const int lane = threadIdx.x & 31;
  const int i    = blockIdx.x * EWAVES + wave;   // this wave owns node i

  h16* Ein  = &sEin[wave][0][0];
  h16* Abuf = &sA  [wave][0][0];
  h16* Bbuf = &sB  [wave][0][0];

  if (lane < 12) sXi[wave][lane] = x[i*12 + lane];
  // zero pad columns 4..7 and 168..191 (paired We1t rows are zero; keep LDS NaN-free)
  for (int idx = lane; idx < 16*4; idx += 32)
    Ein[(idx >> 2)*FEP + 4 + (idx & 3)] = (h16)0.f;
  for (int idx = lane; idx < 16*24; idx += 32)
    Ein[(idx / 24)*FEP + 168 + (idx % 24)] = (h16)0.f;
  // static feat_i block (cols 88..167), vectorized 16B copies
  for (int idx = lane; idx < 160; idx += 32){
    int m = idx / 10, cb = idx - m*10;
    *(v8h*)(Ein + m*FEP + 88 + cb*8) = *(const v8h*)(feat + i*80 + cb*8);
  }
  __syncthreads();

  float accMi[4] = {0.f,0.f,0.f,0.f};
  float accS = 0.f;
  const int hh = lane / 3, dd = lane - hh*3;   // lanes 0..11 own (head,dim)
  const float binf0 = binf[0];

  for (int jt = 0; jt < NNODE/16; ++jt){
    const int jb = jt*16;
    // P0: pair squared norms (cols 0..3) + feat_j block (cols 8..87, 16B copies)
    for (int idx = lane; idx < 64; idx += 32){
      int m = idx >> 2, c = idx & 3;
      int j = jb + m;
      float d0 = x[j*12 + c*3 + 0] - sXi[wave][c*3 + 0];
      float d1 = x[j*12 + c*3 + 1] - sXi[wave][c*3 + 1];
      float d2 = x[j*12 + c*3 + 2] - sXi[wave][c*3 + 2];
      float sq = d0*d0 + d1*d1 + d2*d2;
      sSq[wave][m][c] = sq;
      Ein[m*FEP + c] = (h16)sq;
    }
    for (int idx = lane; idx < 160; idx += 32){
      int m = idx / 10, cb = idx - m*10;
      *(v8h*)(Ein + m*FEP + 8 + cb*8) = *(const v8h*)(feat + (jb + m)*80 + cb*8);
    }
    __syncthreads();
    mlp_layer(Ein,  FEP, FEP, We1t, be1, Abuf, lane, true);   // L1 silu
    __syncthreads();
    mlp_layer(Abuf, MDIM, MDIM, We2t, be2, Bbuf, lane, true); // L2 silu -> m_ij
    __syncthreads();
    { // gate: e = sigmoid(m_ij @ Winf + binf), diag masked
      v8f c = head_gemm(Bbuf, MDIM, MDIM, Winft, 0, lane);
      if ((lane & 15) == 0){
        int mb = (lane & 16) ? 8 : 0;
        #pragma unroll
        for (int r = 0; r < 8; ++r){
          int m = mb + r;
          sE[wave][m] = (jb + m == i) ? 0.f : sigm(c[r] + binf0);
        }
      }
    }
    __syncthreads();
    // m_i += sum_j e * m_ij : each lane owns 4 contiguous columns -> b64 LDS reads
    #pragma unroll
    for (int m = 0; m < 16; ++m){
      float e = sE[wave][m];
      v4h v = *(const v4h*)(Bbuf + m*MDIM + lane*4);
      accMi[0] += e * (float)v[0];
      accMi[1] += e * (float)v[1];
      accMi[2] += e * (float)v[2];
      accMi[3] += e * (float)v[3];
    }
    mlp_layer(Bbuf, MDIM, MDIM, Wx1t, bx1, Abuf, lane, true); // L3 silu
    __syncthreads();
    mlp_layer(Abuf, MDIM, MDIM, Wx2t, bx2, Bbuf, lane, true); // L4 silu
    __syncthreads();
    { // phi_x heads: [16][4]
      v8f c = head_gemm(Bbuf, MDIM, MDIM, Wx3t, 0, lane);
      int nn = lane & 15;
      if (nn < 4){
        int mb = (lane & 16) ? 8 : 0;
        #pragma unroll
        for (int r = 0; r < 8; ++r) sPx[wave][mb + r][nn] = c[r] + bx3[nn];
      }
    }
    __syncthreads();
    // equivariant shift accumulation (diag contributes 0 since diff==0)
    if (lane < 12){
      #pragma unroll
      for (int m = 0; m < 16; ++m){
        int j = jb + m;
        float inv  = fastrcp(__builtin_amdgcn_sqrtf(sSq[wave][m][hh] + 1e-8f) + 1.f);
        float diff = x[j*12 + hh*3 + dd] - sXi[wave][hh*3 + dd];
        accS += diff * inv * sPx[wave][m][hh];
      }
    }
    __syncthreads();
  }
  { // contiguous 16B store of this lane's 4 m_i columns
    float4 o; o.x = accMi[0]; o.y = accMi[1]; o.z = accMi[2]; o.w = accMi[3];
    *(float4*)(mi + (size_t)i*MDIM + lane*4) = o;
  }
  if (lane < 12) Sx[i*12 + lane] = accS;
}

// ---------------- pass 2: per-node MLPs + outputs ----------------
#define NWAVES 4
__global__ void __launch_bounds__(NWAVES*32) egcl_node(
    const float* __restrict__ x, const float* __restrict__ h,
    const float* __restrict__ mi, const float* __restrict__ Sx,
    const h16* __restrict__ Wc1t, const h16* __restrict__ Wc2t,
    const h16* __restrict__ Wc3t, const h16* __restrict__ Wh1t,
    const h16* __restrict__ Wh2t, const h16* __restrict__ Wh3t,
    const float* __restrict__ bc1, const float* __restrict__ bc2,
    const float* __restrict__ bc3, const float* __restrict__ bh1,
    const float* __restrict__ bh2, const float* __restrict__ bh3,
    float* __restrict__ outX, float* __restrict__ outH)
{
  __shared__ __align__(32) h16 sAin[NWAVES][16][PHK];
  __shared__ __align__(32) h16 sB1 [NWAVES][16][MDIM];
  __shared__ __align__(32) h16 sB2 [NWAVES][16][MDIM];
  __shared__ float sPxc[NWAVES][16][16];

  const int wave = threadIdx.x >> 5;
  const int lane = threadIdx.x & 31;
  const int nb   = (blockIdx.x * NWAVES + wave) * 16;

  h16* Ain = &sAin[wave][0][0];
  h16* B1  = &sB1 [wave][0][0];
  h16* B2  = &sB2 [wave][0][0];

  for (int idx = lane; idx < 16*PHK; idx += 32){
    int m = idx / PHK, c = idx % PHK;
    int node = nb + m;
    Ain[idx] = (c < MDIM) ? (h16)mi[node*MDIM + c] : (h16)h[node*DHDIM + (c - MDIM)];
  }
  __syncthreads();
  mlp_layer(Ain, PHK, MDIM, Wc1t, bc1, B1, lane, true);   // phi_c L1 (m_i part only)
  __syncthreads();
  mlp_layer(B1, MDIM, MDIM, Wc2t, bc2, B2, lane, true);   // phi_c L2
  __syncthreads();
  { // phi_x_cross: [16 nodes][16 = a*4+b]
    v8f c = head_gemm(B2, MDIM, MDIM, Wc3t, 0, lane);
    int nn = lane & 15, mb = (lane & 16) ? 8 : 0;
    #pragma unroll
    for (int r = 0; r < 8; ++r) sPxc[wave][mb + r][nn] = c[r] + bc3[nn];
  }
  __syncthreads();
  mlp_layer(Ain, PHK, PHK, Wh1t, bh1, B1, lane, true);    // phi_h L1 (full 192)
  __syncthreads();
  mlp_layer(B1, MDIM, MDIM, Wh2t, bh2, B2, lane, false);  // phi_h L2 (no act)
  __syncthreads();
  { // h_new = h + B2 @ Wh3 + bh3
    int nn = lane & 15, mb = (lane & 16) ? 8 : 0;
    for (int nt = 0; nt < 4; ++nt){
      v8f c = head_gemm(B2, MDIM, MDIM, Wh3t, nt*16, lane);
      int col = nt*16 + nn;
      #pragma unroll
      for (int r = 0; r < 8; ++r){
        int node = nb + mb + r;
        outH[node*DHDIM + col] = h[node*DHDIM + col] + c[r] + bh3[col];
      }
    }
  }
  // x_new = x + (Sx + shift_cross)/(N-1); lanes cover (node_local m, head b)
  #pragma unroll
  for (int t = 0; t < 2; ++t){
    int idx = lane + 32*t;
    int m = idx >> 2, b = idx & 3;
    int node = nb + m;
    for (int d = 0; d < 3; ++d){
      float acc = 0.f;
      #pragma unroll
      for (int a = 0; a < 4; ++a){
        float d0 = x[node*12 + b*3 + 0] - x[node*12 + a*3 + 0];
        float d1 = x[node*12 + b*3 + 1] - x[node*12 + a*3 + 1];
        float d2 = x[node*12 + b*3 + 2] - x[node*12 + a*3 + 2];
        float sq = d0*d0 + d1*d1 + d2*d2;
        float inv = fastrcp(__builtin_amdgcn_sqrtf(sq + 1e-8f) + 1.f);
        float dv = (d == 0) ? d0 : ((d == 1) ? d1 : d2);
        acc += dv * inv * sPxc[wave][m][a*4 + b];   // diag term: dv==0
      }
      int o = node*12 + b*3 + d;
      outX[o] = x[o] + (Sx[o] + acc) * (1.f/767.f);
    }
  }
}

extern "C" void kernel_launch(void* const* d_in, const int* in_sizes, int n_in,
                              void* d_out, int out_size, void* d_ws, size_t ws_size,
                              hipStream_t stream) {
  (void)in_sizes; (void)n_in; (void)out_size; (void)ws_size;
  const float* x   = (const float*)d_in[0];
  const float* h   = (const float*)d_in[1];
  const float* We1 = (const float*)d_in[2];  const float* be1 = (const float*)d_in[3];
  const float* We2 = (const float*)d_in[4];  const float* be2 = (const float*)d_in[5];
  const float* Winf= (const float*)d_in[6];  const float* binf= (const float*)d_in[7];
  const float* Wx1 = (const float*)d_in[8];  const float* bx1 = (const float*)d_in[9];
  const float* Wx2 = (const float*)d_in[10]; const float* bx2 = (const float*)d_in[11];
  const float* Wx3 = (const float*)d_in[12]; const float* bx3 = (const float*)d_in[13];
  const float* Wc1 = (const float*)d_in[14]; const float* bc1 = (const float*)d_in[15];
  const float* Wc2 = (const float*)d_in[16]; const float* bc2 = (const float*)d_in[17];
  const float* Wc3 = (const float*)d_in[18]; const float* bc3 = (const float*)d_in[19];
  const float* Wh1 = (const float*)d_in[20]; const float* bh1 = (const float*)d_in[21];
  const float* Wh2 = (const float*)d_in[22]; const float* bh2 = (const float*)d_in[23];
  const float* Wh3 = (const float*)d_in[24]; const float* bh3 = (const float*)d_in[25];

  char* ws = (char*)d_ws;
  size_t off = 0;
  auto take = [&](size_t bytes)->char* {
    char* p = ws + off; off += (bytes + 255) & ~(size_t)255; return p;
  };
  h16* feat  = (h16*)take((size_t)NNODE*80*2);
  h16* We1t  = (h16*)take(128*192*2);
  h16* We2t  = (h16*)take(128*128*2);
  h16* Wx1t  = (h16*)take(128*128*2);
  h16* Wx2t  = (h16*)take(128*128*2);
  h16* Winft = (h16*)take(16*128*2);
  h16* Wx3t  = (h16*)take(16*128*2);
  h16* Wc1t  = (h16*)take(128*128*2);
  h16* Wc2t  = (h16*)take(128*128*2);
  h16* Wc3t  = (h16*)take(16*128*2);
  h16* Wh1t  = (h16*)take(128*192*2);
  h16* Wh2t  = (h16*)take(128*128*2);
  h16* Wh3t  = (h16*)take(64*128*2);
  float* mi  = (float*)take((size_t)NNODE*MDIM*4);
  float* Sx  = (float*)take((size_t)NNODE*12*4);

  float* outX = (float*)d_out;
  float* outH = outX + (size_t)NNODE*NHEAD*3;

  egcl_prep<<<128, 256, 0, stream>>>(x, h, We1, We2, Winf, Wx1, Wx2, Wx3,
                                     Wc1, Wc2, Wc3, Wh1, Wh2, Wh3,
                                     feat, We1t, We2t, Winft, Wx1t, Wx2t, Wx3t,
                                     Wc1t, Wc2t, Wc3t, Wh1t, Wh2t, Wh3t);
  egcl_edge<<<NNODE/EWAVES, EWAVES*32, 0, stream>>>(x, feat, We1t, We2t, Winft,
                                     Wx1t, Wx2t, Wx3t, be1, be2, binf, bx1, bx2, bx3,
                                     mi, Sx);
  egcl_node<<<NNODE/(NWAVES*16), NWAVES*32, 0, stream>>>(x, h, mi, Sx,
                                     Wc1t, Wc2t, Wc3t, Wh1t, Wh2t, Wh3t,
                                     bc1, bc2, bc3, bh1, bh2, bh3, outX, outH);
}
